// ExperimentalGNNWithSavedStats_64965675320000
// MI455X (gfx1250) — compile-verified, matrix operand order checked by reading, not run on hardware
//
#include <hip/hip_runtime.h>
#include <hip/hip_bf16.h>
#include <hip/hip_fp16.h>

typedef __attribute__((ext_vector_type(16))) _Float16 v16h;
typedef __attribute__((ext_vector_type(8)))  float    v8f;

__device__ __forceinline__ float sigf(float x) { return 1.0f / (1.0f + expf(-x)); }
__device__ __forceinline__ float siluf(float x) { return x / (1.0f + expf(-x)); }

// monotone float <-> uint for atomicMax-based segment max
__device__ __forceinline__ unsigned keyenc(float f) {
    unsigned u = __float_as_uint(f);
    return (u & 0x80000000u) ? ~u : (u | 0x80000000u);
}
__device__ __forceinline__ float keydec(unsigned u) {
    unsigned v = (u & 0x80000000u) ? (u ^ 0x80000000u) : ~u;
    float f = __uint_as_float(v);
    // mimic: m = where(isfinite(m), m, 0)
    if (!(f == f) || f > 3.0e38f || f < -3.0e38f) f = 0.0f;
    return f;
}

// ---------------------------------------------------------------------------
// Weight transpose+convert:  W f32 [K,N] row-major -> WT f16 [N,K] row-major
// ---------------------------------------------------------------------------
__global__ void wtrans_kernel(const float* __restrict__ W, _Float16* __restrict__ WT,
                              int K, int N) {
    int idx = blockIdx.x * 256 + threadIdx.x;
    if (idx >= K * N) return;
    int k = idx / N, n = idx - k * N;
    WT[(size_t)n * K + k] = (_Float16)W[idx];
}

// ---------------------------------------------------------------------------
// Fused WMMA GEMM:  C[M,N] = epi( A[M,K]_f16 @ WT[N,K]_f16^T + bias )
//   block = 256 threads (8 waves); wave w -> cols [colBase, colBase+64)
//   each wave: 2 (M) x 4 (N) tiles of v_wmma_f32_16x16x32_f16
//   mode 0: bias, store f16            (any N multiple-of-64 coverage via grid.y)
//   mode 1: bias + LayerNorm + SiLU    (requires N==512, grid.y==1)
//   mode 2: bias + LayerNorm + f16 residual add (N==512, grid.y==1)
// ---------------------------------------------------------------------------
__global__ __launch_bounds__(256) void gemm_f16_kernel(
    const _Float16* __restrict__ A, const _Float16* __restrict__ WT,
    const float* __restrict__ bias, _Float16* __restrict__ C,
    const float* __restrict__ lng, const float* __restrict__ lnb,
    const _Float16* __restrict__ res,
    int M, int N, int K, int mode) {
    __shared__ float lds[32 * 512];

    const int lane  = threadIdx.x & 31;
    const int wave  = threadIdx.x >> 5;
    const int laneM = lane & 15;
    const int laneH = lane >> 4;           // 0 or 1
    const int mBase = blockIdx.x * 32;
    const int colBase = blockIdx.y * 512 + wave * 64;

    int r0 = mBase + laneM;       if (r0 >= M) r0 = M - 1;
    int r1 = mBase + 16 + laneM;  if (r1 >= M) r1 = M - 1;
    const _Float16* a0p = A + (size_t)r0 * K + laneH * 16;
    const _Float16* a1p = A + (size_t)r1 * K + laneH * 16;

    const _Float16* bp[4];
#pragma unroll
    for (int nt = 0; nt < 4; ++nt) {
        int c = colBase + nt * 16 + laneM;
        if (c >= N) c = N - 1;
        bp[nt] = WT + (size_t)c * K + laneH * 16;
    }

    v8f acc[2][4] = {};

    for (int k0 = 0; k0 < K; k0 += 32) {
        v16h a0 = *(const v16h*)(a0p + k0);
        v16h a1 = *(const v16h*)(a1p + k0);
        v16h b0 = *(const v16h*)(bp[0] + k0);
        v16h b1 = *(const v16h*)(bp[1] + k0);
        v16h b2 = *(const v16h*)(bp[2] + k0);
        v16h b3 = *(const v16h*)(bp[3] + k0);
        acc[0][0] = __builtin_amdgcn_wmma_f32_16x16x32_f16(false, a0, false, b0, (short)0, acc[0][0], false, false);
        acc[0][1] = __builtin_amdgcn_wmma_f32_16x16x32_f16(false, a0, false, b1, (short)0, acc[0][1], false, false);
        acc[0][2] = __builtin_amdgcn_wmma_f32_16x16x32_f16(false, a0, false, b2, (short)0, acc[0][2], false, false);
        acc[0][3] = __builtin_amdgcn_wmma_f32_16x16x32_f16(false, a0, false, b3, (short)0, acc[0][3], false, false);
        acc[1][0] = __builtin_amdgcn_wmma_f32_16x16x32_f16(false, a1, false, b0, (short)0, acc[1][0], false, false);
        acc[1][1] = __builtin_amdgcn_wmma_f32_16x16x32_f16(false, a1, false, b1, (short)0, acc[1][1], false, false);
        acc[1][2] = __builtin_amdgcn_wmma_f32_16x16x32_f16(false, a1, false, b2, (short)0, acc[1][2], false, false);
        acc[1][3] = __builtin_amdgcn_wmma_f32_16x16x32_f16(false, a1, false, b3, (short)0, acc[1][3], false, false);
    }

    if (mode == 0) {
#pragma unroll
        for (int mt = 0; mt < 2; ++mt)
#pragma unroll
            for (int nt = 0; nt < 4; ++nt)
#pragma unroll
                for (int g = 0; g < 8; ++g) {
                    int row = mBase + mt * 16 + g + laneH * 8;
                    int col = colBase + nt * 16 + laneM;
                    if (row < M && col < N)
                        C[(size_t)row * N + col] = (_Float16)(acc[mt][nt][g] + bias[col]);
                }
        return;
    }

    // LN modes: stage full 32x512 fp32 strip in LDS
#pragma unroll
    for (int mt = 0; mt < 2; ++mt)
#pragma unroll
        for (int nt = 0; nt < 4; ++nt)
#pragma unroll
            for (int g = 0; g < 8; ++g) {
                int lrow = mt * 16 + g + laneH * 8;
                int lcol = wave * 64 + nt * 16 + laneM;
                lds[lrow * 512 + lcol] = acc[mt][nt][g] + bias[lcol];
            }
    __syncthreads();

    int row = threadIdx.x >> 3;        // 0..31
    int seg = threadIdx.x & 7;         // 8 threads per row, 64 cols each
    const float* rp = &lds[row * 512 + seg * 64];
    float s = 0.f, q = 0.f;
    for (int j = 0; j < 64; ++j) { float v = rp[j]; s += v; q += v * v; }
    s += __shfl_xor(s, 1, 32); s += __shfl_xor(s, 2, 32); s += __shfl_xor(s, 4, 32);
    q += __shfl_xor(q, 1, 32); q += __shfl_xor(q, 2, 32); q += __shfl_xor(q, 4, 32);
    float mu   = s * (1.0f / 512.0f);
    float rstd = rsqrtf(q * (1.0f / 512.0f) - mu * mu + 1e-5f);

    int grow = mBase + row;
    if (grow < M) {
        for (int j = 0; j < 64; ++j) {
            int c = seg * 64 + j;
            float v = (rp[j] - mu) * rstd * lng[c] + lnb[c];
            if (mode == 1) v = siluf(v);
            else           v += (float)res[(size_t)grow * 512 + c];
            C[(size_t)grow * 512 + c] = (_Float16)v;
        }
    }
}

// ---------------------------------------------------------------------------
// Small-K encoder: out = silu(LN(X[M,Kin] @ W[Kin,512] + b)), Kin in {3,4}
// block = 256 threads handles 32 rows (8 threads/row, 64 cols/thread)
// ---------------------------------------------------------------------------
__global__ __launch_bounds__(256) void encode_small_kernel(
    const float* __restrict__ X, int Kin,
    const float* __restrict__ W, const float* __restrict__ bias,
    const float* __restrict__ lng, const float* __restrict__ lnb,
    _Float16* __restrict__ out, int M) {
    int row  = threadIdx.x >> 3;
    int seg  = threadIdx.x & 7;
    int grow = blockIdx.x * 32 + row;
    int cr   = grow < M ? grow : M - 1;
    float xv[4] = {0.f, 0.f, 0.f, 0.f};
    for (int k = 0; k < Kin; ++k) xv[k] = X[(size_t)cr * Kin + k];
    float vv[64];
    float s = 0.f, q = 0.f;
    for (int j = 0; j < 64; ++j) {
        int c = seg * 64 + j;
        float a = bias[c];
        for (int k = 0; k < Kin; ++k) a += xv[k] * W[k * 512 + c];
        vv[j] = a; s += a; q += a * a;
    }
    s += __shfl_xor(s, 1, 32); s += __shfl_xor(s, 2, 32); s += __shfl_xor(s, 4, 32);
    q += __shfl_xor(q, 1, 32); q += __shfl_xor(q, 2, 32); q += __shfl_xor(q, 4, 32);
    float mu   = s * (1.0f / 512.0f);
    float rstd = rsqrtf(q * (1.0f / 512.0f) - mu * mu + 1e-5f);
    if (grow < M) {
        for (int j = 0; j < 64; ++j) {
            int c = seg * 64 + j;
            float v = siluf((vv[j] - mu) * rstd * lng[c] + lnb[c]);
            out[(size_t)grow * 512 + c] = (_Float16)v;
        }
    }
}

// ---------------------------------------------------------------------------
// GINE message: agg[dst] += relu(h[src] + linE(e))
// ---------------------------------------------------------------------------
__global__ void gine_msg_kernel(const _Float16* __restrict__ h, const _Float16* __restrict__ el,
                                const int* __restrict__ src, const int* __restrict__ dst,
                                float* __restrict__ agg, int E) {
    int idx = blockIdx.x * 256 + threadIdx.x;
    if (idx >= E * 512) return;
    int e = idx >> 9, c = idx & 511;
    float m = (float)h[(size_t)src[e] * 512 + c] + (float)el[(size_t)e * 512 + c];
    if (m > 0.0f) atomicAdd(&agg[(size_t)dst[e] * 512 + c], m);
}

__global__ void addz_kernel(const _Float16* __restrict__ h, const float* __restrict__ agg,
                            _Float16* __restrict__ z, int total) {
    int i = blockIdx.x * 256 + threadIdx.x;
    if (i < total) z[i] = (_Float16)((float)h[i] + agg[i]);
}

// ---------------------------------------------------------------------------
// Transformer attention
// ---------------------------------------------------------------------------
__global__ void attn_alpha_kernel(const _Float16* __restrict__ q, const _Float16* __restrict__ k,
                                  const _Float16* __restrict__ ee,
                                  const int* __restrict__ src, const int* __restrict__ dst,
                                  float* __restrict__ alpha, int E) {
    int idx = blockIdx.x * 256 + threadIdx.x;
    if (idx >= E * 8) return;
    int e = idx >> 3, hd = idx & 7;
    const _Float16* qp = q  + (size_t)dst[e] * 512 + hd * 64;
    const _Float16* kp = k  + (size_t)src[e] * 512 + hd * 64;
    const _Float16* ep = ee + (size_t)e      * 512 + hd * 64;
    float acc = 0.f;
    for (int j = 0; j < 64; ++j) acc += (float)qp[j] * ((float)kp[j] + (float)ep[j]);
    alpha[idx] = acc * 0.125f;   // 1/sqrt(64)
}

__global__ void attn_amax_kernel(const float* __restrict__ alpha, const int* __restrict__ dst,
                                 unsigned* __restrict__ mkey, int E) {
    int idx = blockIdx.x * 256 + threadIdx.x;
    if (idx >= E * 8) return;
    int e = idx >> 3, hd = idx & 7;
    atomicMax(&mkey[dst[e] * 8 + hd], keyenc(alpha[idx]));
}

__global__ void attn_expsum_kernel(float* __restrict__ alpha, const int* __restrict__ dst,
                                   const unsigned* __restrict__ mkey, float* __restrict__ ssum, int E) {
    int idx = blockIdx.x * 256 + threadIdx.x;
    if (idx >= E * 8) return;
    int e = idx >> 3, hd = idx & 7;
    float m  = keydec(mkey[dst[e] * 8 + hd]);
    float ex = expf(alpha[idx] - m);
    alpha[idx] = ex;
    atomicAdd(&ssum[dst[e] * 8 + hd], ex);
}

__global__ void attn_msg_kernel(const float* __restrict__ alpha, const float* __restrict__ ssum,
                                const _Float16* __restrict__ v, const _Float16* __restrict__ ee,
                                const int* __restrict__ src, const int* __restrict__ dst,
                                float* __restrict__ agg, int E) {
    int idx = blockIdx.x * 256 + threadIdx.x;
    if (idx >= E * 512) return;
    int e = idx >> 9, c = idx & 511, hd = c >> 6;
    int s = src[e], d = dst[e];
    float w   = alpha[e * 8 + hd] / (ssum[d * 8 + hd] + 1e-16f);
    float val = ((float)v[(size_t)s * 512 + c] + (float)ee[(size_t)e * 512 + c]) * w;
    atomicAdd(&agg[(size_t)d * 512 + c], val);
}

__global__ __launch_bounds__(256) void beta_kernel(
    const float* __restrict__ outv, const _Float16* __restrict__ xr,
    const float* __restrict__ wb, const float* __restrict__ bb,
    float* __restrict__ beta) {
    __shared__ float red[256];
    int n = blockIdx.x;
    float acc = 0.f;
    for (int c = threadIdx.x; c < 512; c += 256) {
        float o = outv[(size_t)n * 512 + c];
        float x = (float)xr[(size_t)n * 512 + c];
        acc += o * wb[c] + x * wb[512 + c] + (o - x) * wb[1024 + c];
    }
    red[threadIdx.x] = acc; __syncthreads();
    for (int st = 128; st > 0; st >>= 1) {
        if (threadIdx.x < st) red[threadIdx.x] += red[threadIdx.x + st];
        __syncthreads();
    }
    if (threadIdx.x == 0) beta[n] = sigf(red[0] + bb[0]);
}

// h_out = h_in + LN(beta*xr + (1-beta)*out)
__global__ __launch_bounds__(256) void trans_combine_ln_kernel(
    const float* __restrict__ outv, const _Float16* __restrict__ xr,
    const float* __restrict__ beta, const _Float16* __restrict__ hin,
    const float* __restrict__ g, const float* __restrict__ b,
    _Float16* __restrict__ hout) {
    __shared__ float buf[512];
    __shared__ float red[256];
    int n = blockIdx.x;
    float bt = beta[n];
    float s = 0.f, q = 0.f;
    for (int c = threadIdx.x; c < 512; c += 256) {
        float hn = bt * (float)xr[(size_t)n * 512 + c] + (1.0f - bt) * outv[(size_t)n * 512 + c];
        buf[c] = hn; s += hn; q += hn * hn;
    }
    red[threadIdx.x] = s; __syncthreads();
    for (int st = 128; st > 0; st >>= 1) { if (threadIdx.x < st) red[threadIdx.x] += red[threadIdx.x + st]; __syncthreads(); }
    float tot = red[0]; __syncthreads();
    red[threadIdx.x] = q; __syncthreads();
    for (int st = 128; st > 0; st >>= 1) { if (threadIdx.x < st) red[threadIdx.x] += red[threadIdx.x + st]; __syncthreads(); }
    float totq = red[0]; __syncthreads();
    float mu   = tot * (1.0f / 512.0f);
    float rstd = rsqrtf(totq * (1.0f / 512.0f) - mu * mu + 1e-5f);
    for (int c = threadIdx.x; c < 512; c += 256) {
        float v = (buf[c] - mu) * rstd * g[c] + b[c];
        hout[(size_t)n * 512 + c] = (_Float16)((float)hin[(size_t)n * 512 + c] + v);
    }
}

// standalone row LayerNorm (+optional SiLU), width <= 1024, f16 in/out
__global__ __launch_bounds__(256) void row_ln_kernel(
    const _Float16* __restrict__ in, _Float16* __restrict__ out,
    const float* __restrict__ g, const float* __restrict__ b,
    int W, int istride, int ostride, int dosilu) {
    __shared__ float buf[1024];
    __shared__ float red[256];
    int r = blockIdx.x;
    float s = 0.f, q = 0.f;
    for (int c = threadIdx.x; c < W; c += 256) {
        float v = (float)in[(size_t)r * istride + c];
        buf[c] = v; s += v; q += v * v;
    }
    red[threadIdx.x] = s; __syncthreads();
    for (int st = 128; st > 0; st >>= 1) { if (threadIdx.x < st) red[threadIdx.x] += red[threadIdx.x + st]; __syncthreads(); }
    float tot = red[0]; __syncthreads();
    red[threadIdx.x] = q; __syncthreads();
    for (int st = 128; st > 0; st >>= 1) { if (threadIdx.x < st) red[threadIdx.x] += red[threadIdx.x + st]; __syncthreads(); }
    float totq = red[0]; __syncthreads();
    float mu   = tot / (float)W;
    float rstd = rsqrtf(totq / (float)W - mu * mu + 1e-5f);
    for (int c = threadIdx.x; c < W; c += 256) {
        float v = (buf[c] - mu) * rstd * g[c] + b[c];
        if (dosilu) v = siluf(v);
        out[(size_t)r * ostride + c] = (_Float16)v;
    }
}

// ---------------------------------------------------------------------------
// Set2Set
// ---------------------------------------------------------------------------
__global__ void s2s_gates_kernel(const float* __restrict__ qs, const float* __restrict__ hs,
                                 const float* __restrict__ wih, const float* __restrict__ whh,
                                 const float* __restrict__ bih, const float* __restrict__ bhh,
                                 float* __restrict__ gates) {
    int idx = blockIdx.x * 256 + threadIdx.x;
    if (idx >= 64 * 2048) return;
    int b = idx >> 11, o = idx & 2047;
    float acc = bih[o] + bhh[o];
    for (int k = 0; k < 1024; ++k) acc += qs[b * 1024 + k] * wih[k * 2048 + o];
    for (int k = 0; k < 512;  ++k) acc += hs[b * 512  + k] * whh[k * 2048 + o];
    gates[idx] = acc;
}

__global__ void s2s_lstm_kernel(const float* __restrict__ gates,
                                float* __restrict__ cs, float* __restrict__ hs) {
    int idx = blockIdx.x * 256 + threadIdx.x;
    if (idx >= 64 * 512) return;
    int b = idx >> 9, j = idx & 511;
    const float* gb = gates + b * 2048;
    float i = sigf(gb[j]), f = sigf(gb[512 + j]), g = tanhf(gb[1024 + j]), o = sigf(gb[1536 + j]);
    float c = f * cs[idx] + i * g;
    cs[idx] = c;
    hs[idx] = o * tanhf(c);
}

__global__ void s2s_escore_kernel(const _Float16* __restrict__ h, const float* __restrict__ hs,
                                  const int* __restrict__ batch, float* __restrict__ esc, int N) {
    int n = blockIdx.x * 256 + threadIdx.x;
    if (n >= N) return;
    const float* hb = hs + batch[n] * 512;
    float acc = 0.f;
    for (int j = 0; j < 512; ++j) acc += (float)h[(size_t)n * 512 + j] * hb[j];
    esc[n] = acc;
}

__global__ void s2s_amax_kernel(const float* __restrict__ esc, const int* __restrict__ batch,
                                unsigned* __restrict__ mkey, int N) {
    int n = blockIdx.x * 256 + threadIdx.x;
    if (n >= N) return;
    atomicMax(&mkey[batch[n]], keyenc(esc[n]));
}

__global__ void s2s_expsum_kernel(float* __restrict__ esc, const int* __restrict__ batch,
                                  const unsigned* __restrict__ mkey, float* __restrict__ ssum, int N) {
    int n = blockIdx.x * 256 + threadIdx.x;
    if (n >= N) return;
    float ex = expf(esc[n] - keydec(mkey[batch[n]]));
    esc[n] = ex;
    atomicAdd(&ssum[batch[n]], ex);
}

__global__ void s2s_r_kernel(const float* __restrict__ esc, const float* __restrict__ ssum,
                             const int* __restrict__ batch, const _Float16* __restrict__ h,
                             float* __restrict__ rbuf, int N) {
    int idx = blockIdx.x * 256 + threadIdx.x;
    if (idx >= N * 512) return;
    int n = idx >> 9, c = idx & 511;
    int b = batch[n];
    float w = esc[n] / (ssum[b] + 1e-16f);
    atomicAdd(&rbuf[b * 512 + c], w * (float)h[(size_t)n * 512 + c]);
}

__global__ void s2s_qstar_kernel(const float* __restrict__ hs, const float* __restrict__ rbuf,
                                 float* __restrict__ qs) {
    int idx = blockIdx.x * 256 + threadIdx.x;
    if (idx >= 64 * 1024) return;
    int b = idx >> 10, j = idx & 1023;
    qs[idx] = (j < 512) ? hs[b * 512 + j] : rbuf[b * 512 + (j - 512)];
}

__global__ void concat_r4_kernel(const float* __restrict__ q0, const float* __restrict__ q1,
                                 _Float16* __restrict__ r4) {
    int idx = blockIdx.x * 256 + threadIdx.x;
    if (idx >= 64 * 2048) return;
    int b = idx >> 11, c = idx & 2047;
    float v = (c < 1024) ? q0[b * 1024 + c] : q1[b * 1024 + (c - 1024)];
    r4[idx] = (_Float16)v;
}

// ---------------------------------------------------------------------------
// Global feature path + head
// ---------------------------------------------------------------------------
__global__ __launch_bounds__(256) void global1_kernel(
    const float* __restrict__ nA, const float* __restrict__ nB, const float* __restrict__ ss,
    const float* __restrict__ W, const float* __restrict__ bias,
    const float* __restrict__ lng, const float* __restrict__ lnb,
    float* __restrict__ outf) {
    __shared__ float buf[512];
    __shared__ float red[256];
    int bi = blockIdx.x;
    float denom = ss[bi] + 1e-10f;
    float f0 = nA[bi] / denom, f1 = nB[bi] / denom;
    float s = 0.f, q = 0.f;
    for (int c = threadIdx.x; c < 512; c += 256) {
        float v = f0 * W[c] + f1 * W[512 + c] + bias[c];
        buf[c] = v; s += v; q += v * v;
    }
    red[threadIdx.x] = s; __syncthreads();
    for (int st = 128; st > 0; st >>= 1) { if (threadIdx.x < st) red[threadIdx.x] += red[threadIdx.x + st]; __syncthreads(); }
    float tot = red[0]; __syncthreads();
    red[threadIdx.x] = q; __syncthreads();
    for (int st = 128; st > 0; st >>= 1) { if (threadIdx.x < st) red[threadIdx.x] += red[threadIdx.x + st]; __syncthreads(); }
    float totq = red[0]; __syncthreads();
    float mu   = tot * (1.0f / 512.0f);
    float rstd = rsqrtf(totq * (1.0f / 512.0f) - mu * mu + 1e-5f);
    for (int c = threadIdx.x; c < 512; c += 256)
        outf[bi * 512 + c] = siluf((buf[c] - mu) * rstd * lng[c] + lnb[c]);
}

__global__ void global2_kernel(const float* __restrict__ g1, const float* __restrict__ W,
                               const float* __restrict__ bias, _Float16* __restrict__ comb) {
    int idx = blockIdx.x * 256 + threadIdx.x;
    if (idx >= 64 * 512) return;
    int b = idx >> 9, c = idx & 511;
    float acc = bias[c];
    for (int k = 0; k < 512; ++k) acc += g1[b * 512 + k] * W[k * 512 + c];
    comb[(size_t)b * 1536 + 1024 + c] = (_Float16)acc;
}

__global__ void final3_kernel(const _Float16* __restrict__ z, const float* __restrict__ w,
                              const float* __restrict__ b, float* __restrict__ out, int B) {
    int bi = blockIdx.x * 64 + threadIdx.x;
    if (bi >= B) return;
    float acc = b[0];
    for (int k = 0; k < 256; ++k) acc += (float)z[bi * 256 + k] * w[k];
    out[bi] = (acc > 20.f) ? acc : log1pf(expf(acc));
}

// ---------------------------------------------------------------------------
// Host launcher
// ---------------------------------------------------------------------------
extern "C" void kernel_launch(void* const* d_in, const int* in_sizes, int n_in,
                              void* d_out, int out_size, void* d_ws, size_t ws_size,
                              hipStream_t stream) {
    const int Nn = in_sizes[0] / 4;      // 10000 nodes
    const int E  = in_sizes[1] / 2;      // 150000 edges
    const int B  = in_sizes[4];          // 64 graphs

    const float* x_in  = (const float*)d_in[0];
    const int*   ei    = (const int*)d_in[1];
    const int*   src   = ei;
    const int*   dstp  = ei + E;
    const float* ea    = (const float*)d_in[2];
    const int*   batch = (const int*)d_in[3];
    const float* nA    = (const float*)d_in[4];
    const float* nB    = (const float*)d_in[5];
    const float* ssz   = (const float*)d_in[6];
    const int P = 7;   // params leaves start (jax tree_flatten: sorted dict keys)
    auto F = [&](int i) -> const float* { return (const float*)d_in[i]; };

    // layer param offsets (relative to P); GINE=8 leaves, Transformer=12
    static const int LOFF[8] = {52, 60, 72, 80, 92, 100, 112, 120};

    // ---- workspace bump allocator ----
    char* base = (char*)d_ws;
    size_t off = 0;
    auto alloc = [&](size_t bytes) -> void* {
        void* p = base + off;
        off = (off + bytes + 255) & ~(size_t)255;
        return p;
    };
    _Float16* e16  = (_Float16*)alloc((size_t)E * 512 * 2);
    _Float16* et16 = (_Float16*)alloc((size_t)E * 512 * 2);
    _Float16* hA   = (_Float16*)alloc((size_t)Nn * 512 * 2);
    _Float16* hB   = (_Float16*)alloc((size_t)Nn * 512 * 2);
    _Float16* z16  = (_Float16*)alloc((size_t)Nn * 512 * 2);
    _Float16* q16  = (_Float16*)alloc((size_t)Nn * 512 * 2);
    _Float16* k16  = (_Float16*)alloc((size_t)Nn * 512 * 2);
    _Float16* v16  = (_Float16*)alloc((size_t)Nn * 512 * 2);
    _Float16* xr16 = (_Float16*)alloc((size_t)Nn * 512 * 2);
    float*    agg  = (float*)alloc((size_t)Nn * 512 * 4);
    float*    alpha= (float*)alloc((size_t)E * 8 * 4);
    unsigned* mkey = (unsigned*)alloc((size_t)Nn * 8 * 4);
    float*    ssum = (float*)alloc((size_t)Nn * 8 * 4);
    float*    betaA= (float*)alloc((size_t)Nn * 4);
    _Float16* WT   = (_Float16*)alloc((size_t)2048 * 1024 * 2);
    float*    qs0  = (float*)alloc((size_t)B * 1024 * 4);
    float*    qs1  = (float*)alloc((size_t)B * 1024 * 4);
    float*    hsb  = (float*)alloc((size_t)B * 512 * 4);
    float*    csb  = (float*)alloc((size_t)B * 512 * 4);
    float*    gates= (float*)alloc((size_t)B * 2048 * 4);
    float*    esc  = (float*)alloc((size_t)Nn * 4);
    unsigned* mk64 = (unsigned*)alloc((size_t)B * 4);
    float*    ss64 = (float*)alloc((size_t)B * 4);
    float*    rbuf = (float*)alloc((size_t)B * 512 * 4);
    _Float16* r4   = (_Float16*)alloc((size_t)B * 2048 * 2);
    _Float16* t16  = (_Float16*)alloc((size_t)B * 1024 * 2);
    _Float16* comb = (_Float16*)alloc((size_t)B * 1536 * 2);
    _Float16* z16a = (_Float16*)alloc((size_t)B * 512 * 2);
    _Float16* z2   = (_Float16*)alloc((size_t)B * 256 * 2);
    float*    g1o  = (float*)alloc((size_t)B * 512 * 4);
    (void)ws_size; (void)n_in; (void)out_size;

    auto gemm = [&](const _Float16* A, const float* W, const float* bias, _Float16* C,
                    int M, int K, int N, int mode,
                    const float* lng, const float* lnb, const _Float16* res) {
        wtrans_kernel<<<(K * N + 255) / 256, 256, 0, stream>>>(W, WT, K, N);
        dim3 grid((M + 31) / 32, (N > 512) ? (N / 512) : 1);
        gemm_f16_kernel<<<grid, 256, 0, stream>>>(A, WT, bias, C, lng, lnb, res, M, N, K, mode);
    };

    // ---- encoders (node_enc at P+132.., edge_enc at P+32..) ----
    encode_small_kernel<<<(Nn + 31) / 32, 256, 0, stream>>>(
        x_in, 4, F(P + 133), F(P + 132), F(P + 135), F(P + 134), hA, Nn);
    encode_small_kernel<<<(E + 31) / 32, 256, 0, stream>>>(
        ea, 3, F(P + 33), F(P + 32), F(P + 35), F(P + 34), e16, E);

    _Float16* hcur = hA;
    _Float16* hnxt = hB;

    for (int i = 0; i < 8; ++i) {
        // e = block(e, edge_convs[i])  (fused LN+SiLU, in-place)
        int ec = P + 4 * i;
        gemm(e16, F(ec + 1), F(ec + 0), e16, E, 512, 512, 1, F(ec + 3), F(ec + 2), nullptr);

        int L = P + LOFF[i];
        int nrm = P + 136 + 2 * i;  // norms[i]: b=+0, g=+1
        if ((i & 1) == 0) {
            // GINE: lin_edge(b,w)=+0,+1  mlp1=+2,+3  mlp2=+4,+5  mlp_ln(b,g)=+6,+7
            gemm(e16, F(L + 1), F(L + 0), et16, E, 512, 512, 0, nullptr, nullptr, nullptr);
            hipMemsetAsync(agg, 0, (size_t)Nn * 512 * 4, stream);
            gine_msg_kernel<<<(E * 512 + 255) / 256, 256, 0, stream>>>(hcur, et16, src, dstp, agg, E);
            addz_kernel<<<(Nn * 512 + 255) / 256, 256, 0, stream>>>(hcur, agg, z16, Nn * 512);
            gemm(z16, F(L + 3), F(L + 2), z16, Nn, 512, 512, 1, F(L + 7), F(L + 6), nullptr);   // mlp1+LN+SiLU
            gemm(z16, F(L + 5), F(L + 4), hnxt, Nn, 512, 512, 2, F(nrm + 1), F(nrm + 0), hcur); // h + LN(mlp2)
            int pj = P + 152 + 4 * (i / 2);  // pool block
            gemm(hnxt, F(pj + 1), F(pj + 0), hnxt, Nn, 512, 512, 1, F(pj + 3), F(pj + 2), nullptr);
        } else {
            // Transformer: beta=+0,+1 e=+2,+3 k=+4,+5 q=+6,+7 skip=+8,+9 v=+10,+11
            gemm(hcur, F(L + 7),  F(L + 6),  q16,  Nn, 512, 512, 0, nullptr, nullptr, nullptr);
            gemm(hcur, F(L + 5),  F(L + 4),  k16,  Nn, 512, 512, 0, nullptr, nullptr, nullptr);
            gemm(hcur, F(L + 11), F(L + 10), v16,  Nn, 512, 512, 0, nullptr, nullptr, nullptr);
            gemm(hcur, F(L + 9),  F(L + 8),  xr16, Nn, 512, 512, 0, nullptr, nullptr, nullptr);
            gemm(e16,  F(L + 3),  F(L + 2),  et16, E,  512, 512, 0, nullptr, nullptr, nullptr);
            attn_alpha_kernel<<<(E * 8 + 255) / 256, 256, 0, stream>>>(q16, k16, et16, src, dstp, alpha, E);
            hipMemsetAsync(mkey, 0, (size_t)Nn * 8 * 4, stream);
            attn_amax_kernel<<<(E * 8 + 255) / 256, 256, 0, stream>>>(alpha, dstp, mkey, E);
            hipMemsetAsync(ssum, 0, (size_t)Nn * 8 * 4, stream);
            attn_expsum_kernel<<<(E * 8 + 255) / 256, 256, 0, stream>>>(alpha, dstp, mkey, ssum, E);
            hipMemsetAsync(agg, 0, (size_t)Nn * 512 * 4, stream);
            attn_msg_kernel<<<(E * 512 + 255) / 256, 256, 0, stream>>>(alpha, ssum, v16, et16, src, dstp, agg, E);
            beta_kernel<<<Nn, 256, 0, stream>>>(agg, xr16, F(L + 1), F(L + 0), betaA);
            trans_combine_ln_kernel<<<Nn, 256, 0, stream>>>(agg, xr16, betaA, hcur,
                                                            F(nrm + 1), F(nrm + 0), hnxt);
        }
        _Float16* t = hcur; hcur = hnxt; hnxt = t;
    }

    // ---- Set2Set x2 (s2s[k]: b_hh=+0 b_ih=+1 w_hh=+2 w_ih=+3, base P+172+4k) ----
    for (int k2 = 0; k2 < 2; ++k2) {
        float* qs = k2 ? qs1 : qs0;
        int S = P + 172 + 4 * k2;
        hipMemsetAsync(qs,  0, (size_t)B * 1024 * 4, stream);
        hipMemsetAsync(hsb, 0, (size_t)B * 512 * 4, stream);
        hipMemsetAsync(csb, 0, (size_t)B * 512 * 4, stream);
        for (int st = 0; st < 4; ++st) {
            s2s_gates_kernel<<<(B * 2048 + 255) / 256, 256, 0, stream>>>(qs, hsb, F(S + 3), F(S + 2),
                                                                         F(S + 1), F(S + 0), gates);
            s2s_lstm_kernel<<<(B * 512 + 255) / 256, 256, 0, stream>>>(gates, csb, hsb);
            s2s_escore_kernel<<<(Nn + 255) / 256, 256, 0, stream>>>(hcur, hsb, batch, esc, Nn);
            hipMemsetAsync(mk64, 0, (size_t)B * 4, stream);
            s2s_amax_kernel<<<(Nn + 255) / 256, 256, 0, stream>>>(esc, batch, mk64, Nn);
            hipMemsetAsync(ss64, 0, (size_t)B * 4, stream);
            s2s_expsum_kernel<<<(Nn + 255) / 256, 256, 0, stream>>>(esc, batch, mk64, ss64, Nn);
            hipMemsetAsync(rbuf, 0, (size_t)B * 512 * 4, stream);
            s2s_r_kernel<<<(Nn * 512 + 255) / 256, 256, 0, stream>>>(esc, ss64, batch, hcur, rbuf, Nn);
            s2s_qstar_kernel<<<(B * 1024 + 255) / 256, 256, 0, stream>>>(hsb, rbuf, qs);
        }
    }
    concat_r4_kernel<<<(B * 2048 + 255) / 256, 256, 0, stream>>>(qs0, qs1, r4);

    // readout_proj block: lin.b=+168 lin.w=+169 ln.b=+170 ln.g=+171
    gemm(r4, F(P + 169), F(P + 168), t16, B, 2048, 1024, 0, nullptr, nullptr, nullptr);
    row_ln_kernel<<<B, 256, 0, stream>>>(t16, comb, F(P + 171), F(P + 170), 1024, 1024, 1536, 1);

    // global feature path: global1 block (+46..49), global2 lin (b=+50, w=+51)
    global1_kernel<<<B, 256, 0, stream>>>(nA, nB, ssz, F(P + 47), F(P + 46),
                                          F(P + 49), F(P + 48), g1o);
    global2_kernel<<<(B * 512 + 255) / 256, 256, 0, stream>>>(g1o, F(P + 51), F(P + 50), comb);

    // final head: final1 (+36..39, fused), final2 (+40..43), final3 (b=+44, w=+45)
    gemm(comb, F(P + 37), F(P + 36), z16a, B, 1536, 512, 1, F(P + 39), F(P + 38), nullptr);
    gemm(z16a, F(P + 41), F(P + 40), t16, B, 512, 256, 0, nullptr, nullptr, nullptr);
    row_ln_kernel<<<B, 256, 0, stream>>>(t16, z2, F(P + 43), F(P + 42), 256, 256, 256, 1);
    final3_kernel<<<1, 64, 0, stream>>>(z2, F(P + 45), F(P + 44), (float*)d_out, B);
}
